// ContrastiveLoss_33767032881199
// MI455X (gfx1250) — matrix operand order repaired and checked by reading, not compiled, not run on hardware
//
#include <hip/hip_runtime.h>
#include <hip/hip_bf16.h>

// SimCLR NT-Xent contrastive loss, fused (no 8192x8192 sim materialization).
// B=4096 rows per view, D=128, TEMP=0.5 -> multiply sim by 2.0 before exp.

typedef __attribute__((ext_vector_type(16))) _Float16 v16h;
typedef __attribute__((ext_vector_type(8)))  _Float16 v8h;
typedef __attribute__((ext_vector_type(4)))  _Float16 v4h;
typedef __attribute__((ext_vector_type(8)))  float    v8f;

#define CL_BATCH   4096
#define CL_NROWS   8192      // 2*BATCH
#define CL_DIM     128
#define CL_INVT    2.0f      // 1/TEMP
#define CL_JTILES  (CL_NROWS / 16)        // 512 column tiles
#define CL_JPW     (CL_JTILES / 8)        // 64 tiles per wave (8 waves/block)

// ---------------------------------------------------------------------------
// Kernel 1: L2-normalize both views, emit f16 reps [2B x D] and exact f32
// positives[b] = dot(z_i[b], z_j[b]). One wave (32 lanes) per row, 4 f32/lane.
// ---------------------------------------------------------------------------
__global__ __launch_bounds__(256) void nx_normalize(const float* __restrict__ p1,
                                                    const float* __restrict__ p2,
                                                    _Float16* __restrict__ reps_h,
                                                    float* __restrict__ pos) {
  const int lane = threadIdx.x & 31;
  const int wave = threadIdx.x >> 5;
  const int row  = blockIdx.x * 8 + wave;        // 0..BATCH-1

  float4 a = ((const float4*)(p1 + (size_t)row * CL_DIM))[lane];
  float4 b = ((const float4*)(p2 + (size_t)row * CL_DIM))[lane];

  float s1 = a.x*a.x + a.y*a.y + a.z*a.z + a.w*a.w;
  float s2 = b.x*b.x + b.y*b.y + b.z*b.z + b.w*b.w;
  #pragma unroll
  for (int off = 16; off; off >>= 1) {
    s1 += __shfl_xor(s1, off, 32);
    s2 += __shfl_xor(s2, off, 32);
  }
  const float i1 = 1.0f / fmaxf(sqrtf(s1), 1e-12f);
  const float i2 = 1.0f / fmaxf(sqrtf(s2), 1e-12f);

  const float zx = a.x*i1, zy = a.y*i1, zz = a.z*i1, zw = a.w*i1;
  const float wx = b.x*i2, wy = b.y*i2, wz = b.z*i2, ww = b.w*i2;

  // exact f32 positive (numerator path matches reference einsum)
  float d = zx*wx + zy*wy + zz*wz + zw*ww;
  #pragma unroll
  for (int off = 16; off; off >>= 1) d += __shfl_xor(d, off, 32);
  if (lane == 0) pos[row] = d;

  v4h h1; h1[0]=(_Float16)zx; h1[1]=(_Float16)zy; h1[2]=(_Float16)zz; h1[3]=(_Float16)zw;
  v4h h2; h2[0]=(_Float16)wx; h2[1]=(_Float16)wy; h2[2]=(_Float16)wz; h2[3]=(_Float16)ww;
  ((v4h*)(reps_h + (size_t)row            * CL_DIM))[lane] = h1;
  ((v4h*)(reps_h + (size_t)(CL_BATCH+row) * CL_DIM))[lane] = h2;
}

// ---------------------------------------------------------------------------
// Kernel 2: one workgroup per 16-row block (512 blocks). The 8 waves of the
// block each sweep 64 of the 512 column tiles; per-tile 16x16 sim is built
// from 4x v_wmma_f32_16x16x32_f16 over K=128. Fragment layouts (this code's k = hi*16 + s order)
// map to contiguous loads from row-major f16 storage:
//   A (16x32, M=lane&15, hi=lane>>4): halves 0..7  = mem[t*32 + hi*16 .. +8)
//                                     halves 8..15 = mem[t*32 + hi*16 + 8 ..)
//   B (32x16, N=lane&15, hi=lane>>4): halves 0..15 = mem[t*32 + hi*16 .. +16)
// B tiles are double-buffered so the load clause for tile j+1 overlaps tile
// j's WMMAs + exp epilogue. The hot loop is fully uniform (no masking): the
// diagonal's computed value is captured once to LDS and its exp subtracted at
// write-out, matching the reference mask exactly. Per-wave row partials
// combine through LDS ds_add_f32.
// ---------------------------------------------------------------------------
__global__ __launch_bounds__(256) void nx_simrow(const _Float16* __restrict__ reps_h,
                                                 float* __restrict__ rowsum) {
  __shared__ float partial[16];
  __shared__ float diagv[16];
  const int lane  = threadIdx.x & 31;
  // readfirstlane: wave id is uniform per wave -> SGPR, scalarizing loop
  // bounds, the diagonal-tile test, and the scalar part of B addresses.
  const int waveS = __builtin_amdgcn_readfirstlane(threadIdx.x >> 5);  // 0..7
  const int iBlk  = blockIdx.x;                  // 0..511 row block
  const int iBase = iBlk * 16;
  const int col   = lane & 15;
  const int hi    = (lane >> 4) & 1;

  if (threadIdx.x < 16) partial[threadIdx.x] = 0.0f;
  __syncthreads();

  // --- A fragments (shared rows for all 8 waves; register-resident) ---
  const _Float16* arow = reps_h + (size_t)(iBase + col) * CL_DIM;
  v16h afrag[4];
  #pragma unroll
  for (int t = 0; t < 4; ++t) {
    v8h lo = *(const v8h*)(arow + t*32 + hi*16);
    v8h up = *(const v8h*)(arow + t*32 + hi*16 + 8);
    v16h f;
    #pragma unroll
    for (int e = 0; e < 8; ++e) { f[e] = lo[e]; f[e+8] = up[e]; }
    afrag[t] = f;
  }

  float acc[8];
  #pragma unroll
  for (int e = 0; e < 8; ++e) acc[e] = 0.0f;

  const int laneOff = col * CL_DIM + hi * 16;    // per-lane, loop-invariant

  auto loadB = [&](int jBlk, v16h* bf) {
    const _Float16* brow = reps_h + (size_t)jBlk * (16 * CL_DIM) + laneOff;
    #pragma unroll
    for (int t = 0; t < 4; ++t) bf[t] = *(const v16h*)(brow + t*32);
  };

  auto accumTile = [&](int jBlk, const v16h* bf) {
    v8f c = {};
    #pragma unroll
    for (int t = 0; t < 4; ++t)
      c = __builtin_amdgcn_wmma_f32_16x16x32_f16(
              /*neg_a=*/false, afrag[t], /*neg_b=*/false, bf[t],
              /*c_mod=*/(short)0, c, /*reuse_a=*/false, /*reuse_b=*/false);
    // Unmasked accumulate: diagonal contribution removed at write-out.
    #pragma unroll
    for (int e = 0; e < 8; ++e)
      acc[e] += __expf(c[e] * CL_INVT);
    if (jBlk == iBlk) {                           // scalar (SCC) branch, 1/64 iters
      #pragma unroll
      for (int e = 0; e < 8; ++e)
        if ((e + hi*8) == col) diagv[col] = c[e]; // lanes holding sim[i][i]
    }
  };

  const int jBeg = waveS * CL_JPW;
  const int jEnd = jBeg + CL_JPW;                 // CL_JPW = 64, even

  v16h b0[4], b1[4];
  loadB(jBeg, b0);
  for (int j = jBeg; j < jEnd; j += 2) {
    loadB(j + 1, b1);                             // overlaps tile j compute
    accumTile(j, b0);
    if (j + 2 < jEnd) loadB(j + 2, b0);           // overlaps tile j+1 compute
    accumTile(j + 1, b1);
  }

  // reduce over the 16 lanes of each half (column dimension of the tile)
  #pragma unroll
  for (int off = 8; off; off >>= 1)
    #pragma unroll
    for (int e = 0; e < 8; ++e)
      acc[e] += __shfl_xor(acc[e], off, 32);

  if (col == 0) {                                 // lanes 0 (rows 0-7) and 16 (rows 8-15)
    #pragma unroll
    for (int e = 0; e < 8; ++e)
      atomicAdd(&partial[e + hi*8], acc[e]);      // ds_add_f32 across the 8 waves
  }
  __syncthreads();
  if (threadIdx.x < 16) {
    const float full = partial[threadIdx.x];
    rowsum[iBase + threadIdx.x] = full - __expf(diagv[threadIdx.x] * CL_INVT);
  }
}

// ---------------------------------------------------------------------------
// Kernel 3: loss = mean_i [ log(denom_i) - pos_{i mod B} / T ]
// ---------------------------------------------------------------------------
__global__ __launch_bounds__(256) void nx_finalize(const float* __restrict__ rowsum,
                                                   const float* __restrict__ pos,
                                                   float* __restrict__ out) {
  __shared__ float red[8];
  float s = 0.0f;
  for (int i = threadIdx.x; i < CL_NROWS; i += 256) {
    const float p = pos[(i < CL_BATCH) ? i : (i - CL_BATCH)];
    s += __logf(rowsum[i]) - p * CL_INVT;
  }
  #pragma unroll
  for (int off = 16; off; off >>= 1) s += __shfl_xor(s, off, 32);
  const int lane = threadIdx.x & 31, wave = threadIdx.x >> 5;
  if (lane == 0) red[wave] = s;
  __syncthreads();
  if (threadIdx.x == 0) {
    float t = 0.0f;
    #pragma unroll
    for (int w = 0; w < 8; ++w) t += red[w];
    out[0] = t / (float)CL_NROWS;
  }
}

// ---------------------------------------------------------------------------
extern "C" void kernel_launch(void* const* d_in, const int* in_sizes, int n_in,
                              void* d_out, int out_size, void* d_ws, size_t ws_size,
                              hipStream_t stream) {
  (void)in_sizes; (void)n_in; (void)out_size; (void)ws_size;
  const float* p1 = (const float*)d_in[0];
  const float* p2 = (const float*)d_in[1];

  _Float16* reps_h = (_Float16*)d_ws;                               // 2 MB
  float*    rowsum = (float*)((char*)d_ws + (size_t)CL_NROWS * CL_DIM * sizeof(_Float16));
  float*    pos    = rowsum + CL_NROWS;                             // +32KB, +16KB
  float*    out    = (float*)d_out;

  nx_normalize<<<CL_BATCH / 8, 256, 0, stream>>>(p1, p2, reps_h, pos);
  nx_simrow  <<<CL_JTILES, 256, 0, stream>>>(reps_h, rowsum);
  nx_finalize<<<1, 256, 0, stream>>>(rowsum, pos, out);
}
